// MistralAttention_71442486002188
// MI455X (gfx1250) — matrix operand order repaired and checked by reading, not compiled, not run on hardware
//
#include <hip/hip_runtime.h>

#define DEV __device__ __forceinline__

typedef __attribute__((ext_vector_type(8)))  __bf16 v8bf;
typedef __attribute__((ext_vector_type(16))) __bf16 v16bf;
typedef __attribute__((ext_vector_type(8)))  float  v8f;
typedef __attribute__((ext_vector_type(4)))  unsigned v4u;
typedef __attribute__((ext_vector_type(8)))  int      v8i;
typedef __attribute__((ext_vector_type(4)))  int      v4i;

#define C_HIDDEN 4096
#define C_NH     32
#define C_NKV    8
#define C_HD     128
#define C_KV     1024
#define C_R      16
#define C_B      2
#define C_S      2048
#define C_WIN    1024
#define C_M      (C_B * C_S)
#define LORA_SCALE 2.0f
#define INV_SQRT_HD 0.08838834764831845f

// ---- CDNA5 async / TDM feature detection (compile-safe fallbacks) ------------
#if __has_builtin(__builtin_amdgcn_global_load_async_to_lds_b128) && \
    __has_builtin(__builtin_amdgcn_s_wait_asynccnt)
#define USE_ASYNC_LDS 1
#endif

#if (__has_builtin(__builtin_amdgcn_tensor_load_to_lds_d2) || \
     __has_builtin(__builtin_amdgcn_tensor_load_to_lds)) && \
    __has_builtin(__builtin_amdgcn_s_wait_tensorcnt)
#define USE_TDM 1
#endif

#define AS3 __attribute__((address_space(3)))
#define AS1 __attribute__((address_space(1)))

#ifdef USE_ASYNC_LDS
// Probe-learned signature: arg0 = AS1(global) v4i*, arg1 = AS3(LDS) v4i*, imm, imm.
DEV void async_copy_b128(void* lds, const void* glob) {
  __builtin_amdgcn_global_load_async_to_lds_b128(
      (AS1 v4i*)glob, (AS3 v4i*)lds, 0, 0);
}
#endif

#ifdef USE_TDM
DEV unsigned lds_byte_off(void* p) {
  return (unsigned)(uintptr_t)(AS3 char*)p;   // addrspacecast strips the aperture
}
// Issue a 2D TDM tile load: group1 prebuilt, group0 per call.
DEV void tdm_load_2d(v4u g0, v8i g1) {
#if __has_builtin(__builtin_amdgcn_tensor_load_to_lds_d2)
  __builtin_amdgcn_tensor_load_to_lds_d2(g0, g1, 0);
#else
  v4i z4 = {};
#if __clang_major__ >= 23
  v8i z8 = {};
  __builtin_amdgcn_tensor_load_to_lds(g0, g1, z4, z4, z8, 0);
#else
  __builtin_amdgcn_tensor_load_to_lds(g0, g1, z4, z4, 0);
#endif
#endif
}
#endif

// ---------------- bf16 helpers (explicit RNE) ---------------------------------
DEV __bf16 f2bf(float f) {
  unsigned u = __builtin_bit_cast(unsigned, f);
  unsigned r = u + 0x7fffu + ((u >> 16) & 1u);
  unsigned short h = (unsigned short)(r >> 16);
  return __builtin_bit_cast(__bf16, h);
}
DEV float bf2f(__bf16 b) {
  unsigned short h = __builtin_bit_cast(unsigned short, b);
  unsigned u = ((unsigned)h) << 16;
  return __builtin_bit_cast(float, u);
}

// A-fragment (16x32 bf16): lane-half holds 16 contiguous K values
DEV v16bf frag_a(const __bf16* p) {
  v8bf lo = *(const v8bf*)(p);
  v8bf hi = *(const v8bf*)(p + 8);
  return __builtin_shufflevector(lo, hi, 0,1,2,3,4,5,6,7,8,9,10,11,12,13,14,15);
}
// B-fragment (32x16 bf16): lane-half holds 16 contiguous K values
DEV v16bf frag_b(const __bf16* p) {
  v8bf lo = *(const v8bf*)(p);
  v8bf hi = *(const v8bf*)(p + 8);
  return __builtin_shufflevector(lo, hi, 0,1,2,3,4,5,6,7,8,9,10,11,12,13,14,15);
}
DEV v8f wmma_bf16(v16bf a, v16bf b, v8f c) {
  return __builtin_amdgcn_wmma_f32_16x16x32_bf16(false, a, false, b,
                                                 (short)0, c, false, false);
}

// ---------------- fp32 -> bf16 conversion (vector x4) --------------------------
__global__ void k_f32_to_bf16(const float* __restrict__ in,
                              __bf16* __restrict__ out, int n4) {
  int i = blockIdx.x * blockDim.x + threadIdx.x;
  if (i >= n4) return;
  float4 v = ((const float4*)in)[i];
  __bf16* o = out + (size_t)i * 4;
  o[0] = f2bf(v.x); o[1] = f2bf(v.y); o[2] = f2bf(v.z); o[3] = f2bf(v.w);
}

// ---------------- LoRA mid: T[M,16] = X[M,K] @ A[K,16] (fp32) -----------------
__global__ void k_lora_mid(const float* __restrict__ X, const float* __restrict__ A,
                           float* __restrict__ T, int Mdim, int Kdim) {
  int idx = blockIdx.x * blockDim.x + threadIdx.x;
  if (idx >= Mdim * C_R) return;
  int row = idx >> 4, col = idx & 15;
  const float* xp = X + (size_t)row * Kdim;
  float acc = 0.f;
  for (int k = 0; k < Kdim; k += 4) {
    acc += xp[k+0] * A[(k+0)*C_R + col];
    acc += xp[k+1] * A[(k+1)*C_R + col];
    acc += xp[k+2] * A[(k+2)*C_R + col];
    acc += xp[k+3] * A[(k+3)*C_R + col];
  }
  T[idx] = acc;
}

// ---------------- Tiled WMMA GEMM: C = A[M,K] @ W[K,N] (+ LoRA epilogue) ------
// Block tile 128x64, 8 waves, wave tile 32x32 (2x2 accumulators), K-step 32.
// A tile staged by the Tensor Data Mover when available.
#define GM_BM 128
#define GM_BN 64
#define GM_BK 32

__global__ __launch_bounds__(256)
void k_gemm_bf16(const __bf16* __restrict__ A, const __bf16* __restrict__ W,
                 const float* __restrict__ T, const float* __restrict__ LB,
                 float* outF, __bf16* outB, int Mdim, int Ndim, int Kdim) {
  __shared__ __align__(16) __bf16 ldsA[GM_BM * GM_BK];   // [row][k]
  __shared__ __align__(16) __bf16 ldsBt[GM_BN * GM_BK];  // [n][k] (transposed W tile)

  const int tid  = threadIdx.x;
  const int lane = tid & 31;
  const int wave = tid >> 5;
  const int lh   = lane >> 4;
  const int lm   = lane & 15;
  const int wm   = wave >> 1;      // 0..3
  const int wn   = wave & 1;       // 0..1

  const int m0 = blockIdx.y * GM_BM;
  const int n0 = blockIdx.x * GM_BN;

  v8f acc[2][2] = {};

  const int ra = tid >> 1;              // 0..127
  const int ka = (tid & 1) * 16;        // 0 / 16
  const int kb = tid >> 3;              // 0..31
  const int nb = (tid & 7) * 8;         // 0..56

#ifdef USE_TDM
  // D# group1 (uniform across loop): data_size=2B, tensor K x M, tile 32x128,
  // dim0 stride = Kdim elements. Bit layout per CDNA5 ISA 8.4.
  v8i g1;
  g1[0] = (int)(1u << 16);                                          // data_size=1 (2B)
  g1[1] = (int)(((unsigned)Kdim & 0xFFFFu) << 16);                  // tensor_dim0 lo16
  g1[2] = (int)((((unsigned)Kdim >> 16) & 0xFFFFu) |
                (((unsigned)Mdim & 0xFFFFu) << 16));                // dim0 hi | dim1 lo
  g1[3] = (int)((((unsigned)Mdim >> 16) & 0xFFFFu) |
                ((unsigned)GM_BK << 16));                           // dim1 hi | tile0=32
  g1[4] = (int)(unsigned)GM_BM;                                     // tile1=128, tile2=0
  g1[5] = (int)(unsigned)Kdim;                                      // stride0 lo32
  g1[6] = 0;                                                        // stride0 hi | stride1 lo
  g1[7] = 0;
  const unsigned ldsA_off = lds_byte_off(&ldsA[0]);
  const unsigned long long Abase =
      (unsigned long long)(uintptr_t)A + ((unsigned long long)m0 * (unsigned)Kdim) * 2ull;
#endif

  for (int k0 = 0; k0 < Kdim; k0 += GM_BK) {
#ifdef USE_TDM
    if (wave == 0) {
      const unsigned long long ga = Abase + (unsigned long long)k0 * 2ull;
      v4u g0;
      g0[0] = 1u;                                                   // count=1
      g0[1] = ldsA_off;                                             // lds_addr
      g0[2] = (unsigned)ga;                                         // global_addr lo
      g0[3] = (unsigned)((ga >> 32) & 0x01FFFFFFu) | (2u << 30);    // addr hi | type=2
      tdm_load_2d(g0, g1);
    }
#else
    // stage A (128x32 row-major)
    const __bf16* ga = A + (size_t)(m0 + ra) * Kdim + k0 + ka;
    *(uint4*)(&ldsA[ra * GM_BK + ka])     = *(const uint4*)(ga);
    *(uint4*)(&ldsA[ra * GM_BK + ka + 8]) = *(const uint4*)(ga + 8);
#endif
    // stage W (32x64) transposed into [n][k]
    const __bf16* gw = W + (size_t)(k0 + kb) * Ndim + n0 + nb;
    v8bf wrow = *(const v8bf*)(gw);
#pragma unroll
    for (int i = 0; i < 8; ++i) ldsBt[(nb + i) * GM_BK + kb] = wrow[i];

    if (k0 + GM_BK < Kdim) {   // prefetch next K-slab -> global_prefetch_b8
#ifndef USE_TDM
      __builtin_prefetch(ga + GM_BK, 0, 1);
#endif
      __builtin_prefetch(gw + (size_t)GM_BK * Ndim, 0, 1);
    }
#ifdef USE_TDM
    if (wave == 0) __builtin_amdgcn_s_wait_tensorcnt(0);
#endif
    __syncthreads();

    v16bf bfr[2];
#pragma unroll
    for (int j = 0; j < 2; ++j)
      bfr[j] = frag_b(&ldsBt[(wn * 32 + j * 16 + lm) * GM_BK + lh * 16]);
#pragma unroll
    for (int i = 0; i < 2; ++i) {
      v16bf af = frag_a(&ldsA[(wm * 32 + i * 16 + lm) * GM_BK + lh * 16]);
#pragma unroll
      for (int j = 0; j < 2; ++j)
        acc[i][j] = wmma_bf16(af, bfr[j], acc[i][j]);
    }
    __syncthreads();
  }

  // epilogue: C/D layout -> element e: row = e + 8*lh, col = lm
#pragma unroll
  for (int j = 0; j < 2; ++j) {
    const int colg = n0 + wn * 32 + j * 16 + lm;
    float lb[C_R];
    if (T) {
#pragma unroll
      for (int r = 0; r < C_R; ++r) lb[r] = LB[r * Ndim + colg];
    }
#pragma unroll
    for (int i = 0; i < 2; ++i) {
#pragma unroll
      for (int e = 0; e < 8; ++e) {
        const int rowg = m0 + wm * 32 + i * 16 + lh * 8 + e;
        float v = acc[i][j][e];
        if (T) {
          float corr = 0.f;
#pragma unroll
          for (int r = 0; r < C_R; ++r) corr += T[rowg * C_R + r] * lb[r];
          v += LORA_SCALE * corr;
        }
        const size_t oidx = (size_t)rowg * Ndim + colg;
        if (outF) outF[oidx] = v;
        else      outB[oidx] = f2bf(v);
      }
    }
  }
}

// ---------------- RoPE (in place on bf16, fp32 math) --------------------------
__global__ void k_rope(__bf16* __restrict__ x, int nheads, int total) {
  int idx = blockIdx.x * blockDim.x + threadIdx.x;
  if (idx >= total) return;
  const int i = idx & 63;
  const int h = (idx >> 6) % nheads;
  const int s = (idx / (64 * nheads)) % C_S;
  const int b = idx / (64 * nheads * C_S);
  const float invf = __expf(-(float)i * 0.14391156847063263f);  // 10000^{-2i/128}
  const float ang = (float)s * invf;
  const float c = cosf(ang), sn = sinf(ang);
  const size_t base = (((size_t)b * C_S + s) * nheads + h) * C_HD;
  const float x1 = bf2f(x[base + i]);
  const float x2 = bf2f(x[base + i + 64]);
  x[base + i]      = f2bf(x1 * c - x2 * sn);
  x[base + i + 64] = f2bf(x2 * c + x1 * sn);
}

// ---------------- Flash attention, sliding window, GQA ------------------------
// grid (S/64, NH, B), 128 threads = 4 waves; each wave owns 16 query rows.
__global__ __launch_bounds__(128)
void k_attn(const __bf16* __restrict__ Q, const __bf16* __restrict__ K,
            const __bf16* __restrict__ V, __bf16* __restrict__ O) {
  __shared__ __align__(16) __bf16 ldsK[32 * C_HD];    // [key][d]  == Bt for QK^T
  __shared__ __align__(16) __bf16 ldsVt[C_HD * 32];   // [d][key]  == Bt for PV
  __shared__ __align__(16) __bf16 ldsP[4 * 16 * 32];  // wave-private P tiles

  const int tid  = threadIdx.x;
  const int lane = tid & 31;
  const int wave = tid >> 5;
  const int lh   = lane >> 4;
  const int lm   = lane & 15;

  const int qt  = blockIdx.x * 64;
  const int h   = blockIdx.y;
  const int b   = blockIdx.z;
  const int kvh = h >> 2;        // N_REP = 4
  const int qbase = qt + wave * 16;

  v16bf qf[4];
  {
    const __bf16* qp = Q + (((size_t)b * C_S + qbase + lm) * C_NH + h) * C_HD;
#pragma unroll
    for (int c = 0; c < 4; ++c) qf[c] = frag_a(qp + c * 32 + lh * 16);
  }

  v8f   oacc[8] = {};
  float mrun[8], lrun[8];
#pragma unroll
  for (int e = 0; e < 8; ++e) { mrun[e] = -3.0e38f; lrun[e] = 0.f; }

  int kt_start = qt - (C_WIN - 1);
  if (kt_start < 0) kt_start = 0;
  kt_start &= ~31;

  for (int kt = kt_start; kt <= qt + 63; kt += 32) {
    __syncthreads();
    {
      const int key = tid >> 2;             // 0..31
      const int d0  = (tid & 3) * 32;       // 0,32,64,96
      const size_t kvrow = ((size_t)b * C_S + kt + key) * C_NKV + kvh;
      const __bf16* kp = K + kvrow * C_HD + d0;
      const __bf16* vp = V + kvrow * C_HD + d0;
#ifdef USE_ASYNC_LDS
#pragma unroll
      for (int i = 0; i < 4; ++i)
        async_copy_b128(&ldsK[key * C_HD + d0 + i * 8], kp + i * 8);
#else
#pragma unroll
      for (int i = 0; i < 4; ++i)
        *(uint4*)(&ldsK[key * C_HD + d0 + i * 8]) = *(const uint4*)(kp + i * 8);
#endif
#pragma unroll
      for (int i = 0; i < 4; ++i) {
        v8bf vv = *(const v8bf*)(vp + i * 8);
#pragma unroll
        for (int u = 0; u < 8; ++u)
          ldsVt[(d0 + i * 8 + u) * 32 + key] = vv[u];
      }
#ifdef USE_ASYNC_LDS
      __builtin_amdgcn_s_wait_asynccnt(0);
#endif
    }
    __syncthreads();

    // scores: 16 rows x 32 keys = two 16x16 accumulators
    v8f s0 = {}, s1 = {};
#pragma unroll
    for (int c = 0; c < 4; ++c) {
      v16bf kf0 = frag_b(&ldsK[(0  + lm) * C_HD + c * 32 + lh * 16]);
      v16bf kf1 = frag_b(&ldsK[(16 + lm) * C_HD + c * 32 + lh * 16]);
      s0 = wmma_bf16(qf[c], kf0, s0);
      s1 = wmma_bf16(qf[c], kf1, s1);
    }

    // online softmax (row stats replicated across each 16-lane group)
    float p0[8], p1[8], sf[8];
#pragma unroll
    for (int e = 0; e < 8; ++e) {
      const int ig = qbase + e + 8 * lh;
      const int j0 = kt + lm, j1 = kt + 16 + lm;
      float x0 = s0[e] * INV_SQRT_HD + ((j0 <= ig && j0 > ig - C_WIN) ? 0.f : -1.0e9f);
      float x1 = s1[e] * INV_SQRT_HD + ((j1 <= ig && j1 > ig - C_WIN) ? 0.f : -1.0e9f);
      float mx = fmaxf(x0, x1);
#pragma unroll
      for (int m = 1; m < 16; m <<= 1) mx = fmaxf(mx, __shfl_xor(mx, m, 32));
      const float mnew = fmaxf(mrun[e], mx);
      const float sc   = __expf(mrun[e] - mnew);
      const float e0   = __expf(x0 - mnew);
      const float e1   = __expf(x1 - mnew);
      float rsum = e0 + e1;
#pragma unroll
      for (int m = 1; m < 16; m <<= 1) rsum += __shfl_xor(rsum, m, 32);
      lrun[e] = lrun[e] * sc + rsum;
      mrun[e] = mnew;
      sf[e] = sc; p0[e] = e0; p1[e] = e1;
    }
#pragma unroll
    for (int j = 0; j < 8; ++j)
#pragma unroll
      for (int e = 0; e < 8; ++e) oacc[j][e] *= sf[e];

    // round-trip P through wave-private LDS to get A-fragment layout
    __bf16* pw = &ldsP[wave * 16 * 32];
#pragma unroll
    for (int e = 0; e < 8; ++e) {
      pw[(e + 8 * lh) * 32 + lm]      = f2bf(p0[e]);
      pw[(e + 8 * lh) * 32 + 16 + lm] = f2bf(p1[e]);
    }
    v16bf pf = frag_a(&pw[lm * 32 + lh * 16]);

    // O += P @ V  (8 x 16-wide d chunks, K=32 keys)
#pragma unroll
    for (int j = 0; j < 8; ++j) {
      v16bf vf = frag_b(&ldsVt[(j * 16 + lm) * 32 + lh * 16]);
      oacc[j] = wmma_bf16(pf, vf, oacc[j]);
    }
  }

  // normalize and store [b][s][h][d]
#pragma unroll
  for (int j = 0; j < 8; ++j) {
    const int d = j * 16 + lm;
#pragma unroll
    for (int e = 0; e < 8; ++e) {
      const int srow = qbase + e + 8 * lh;
      const float v = oacc[j][e] / lrun[e];
      O[(((size_t)b * C_S + srow) * C_NH + h) * C_HD + d] = f2bf(v);
    }
  }
}

// ---------------- host-side orchestration -------------------------------------
extern "C" void kernel_launch(void* const* d_in, const int* in_sizes, int n_in,
                              void* d_out, int out_size, void* d_ws, size_t ws_size,
                              hipStream_t stream) {
  const float* X  = (const float*)d_in[0];
  // d_in[1] = attention_mask (analytic, unused), d_in[2] = position_ids (arange, unused)
  const float* Wq = (const float*)d_in[3];
  const float* Wk = (const float*)d_in[4];
  const float* Wv = (const float*)d_in[5];
  const float* Wo = (const float*)d_in[6];
  const float* Aq = (const float*)d_in[7];
  const float* LBq = (const float*)d_in[8];
  const float* Av = (const float*)d_in[9];
  const float* LBv = (const float*)d_in[10];
  float* out = (float*)d_out;

  char* ws = (char*)d_ws;
  size_t off = 0;
  auto alloc = [&](size_t bytes) -> void* {
    void* p = ws + off;
    off += (bytes + 255) & ~(size_t)255;
    return p;
  };
  __bf16* Xb  = (__bf16*)alloc((size_t)C_M * C_HIDDEN * 2);
  __bf16* Wqb = (__bf16*)alloc((size_t)C_HIDDEN * C_HIDDEN * 2);
  __bf16* Wkb = (__bf16*)alloc((size_t)C_HIDDEN * C_KV * 2);
  __bf16* Wvb = (__bf16*)alloc((size_t)C_HIDDEN * C_KV * 2);
  __bf16* Wob = (__bf16*)alloc((size_t)C_HIDDEN * C_HIDDEN * 2);
  __bf16* Qb  = (__bf16*)alloc((size_t)C_M * C_HIDDEN * 2);
  __bf16* Kb  = (__bf16*)alloc((size_t)C_M * C_KV * 2);
  __bf16* Vb  = (__bf16*)alloc((size_t)C_M * C_KV * 2);
  __bf16* AOb = (__bf16*)alloc((size_t)C_M * C_HIDDEN * 2);
  float*  Tq  = (float*)alloc((size_t)C_M * C_R * 4);
  float*  Tv  = (float*)alloc((size_t)C_M * C_R * 4);

  auto conv = [&](const float* src, __bf16* dst, size_t n) {
    int n4 = (int)(n / 4);
    k_f32_to_bf16<<<(n4 + 255) / 256, 256, 0, stream>>>(src, dst, n4);
  };
  conv(X,  Xb,  (size_t)C_M * C_HIDDEN);
  conv(Wq, Wqb, (size_t)C_HIDDEN * C_HIDDEN);
  conv(Wk, Wkb, (size_t)C_HIDDEN * C_KV);
  conv(Wv, Wvb, (size_t)C_HIDDEN * C_KV);
  conv(Wo, Wob, (size_t)C_HIDDEN * C_HIDDEN);

  k_lora_mid<<<(C_M * C_R + 255) / 256, 256, 0, stream>>>(X, Aq, Tq, C_M, C_HIDDEN);
  k_lora_mid<<<(C_M * C_R + 255) / 256, 256, 0, stream>>>(X, Av, Tv, C_M, C_HIDDEN);

  dim3 blk(256);
  k_gemm_bf16<<<dim3(C_HIDDEN / GM_BN, C_M / GM_BM), blk, 0, stream>>>(
      Xb, Wqb, Tq, LBq, nullptr, Qb, C_M, C_HIDDEN, C_HIDDEN);
  k_gemm_bf16<<<dim3(C_KV / GM_BN, C_M / GM_BM), blk, 0, stream>>>(
      Xb, Wkb, nullptr, nullptr, nullptr, Kb, C_M, C_KV, C_HIDDEN);
  k_gemm_bf16<<<dim3(C_KV / GM_BN, C_M / GM_BM), blk, 0, stream>>>(
      Xb, Wvb, Tv, LBv, nullptr, Vb, C_M, C_KV, C_HIDDEN);

  {
    int tq = C_B * C_S * C_NH * 64;
    k_rope<<<(tq + 255) / 256, 256, 0, stream>>>(Qb, C_NH, tq);
    int tk = C_B * C_S * C_NKV * 64;
    k_rope<<<(tk + 255) / 256, 256, 0, stream>>>(Kb, C_NKV, tk);
  }

  k_attn<<<dim3(C_S / 64, C_NH, C_B), dim3(128), 0, stream>>>(Qb, Kb, Vb, AOb);

  k_gemm_bf16<<<dim3(C_HIDDEN / GM_BN, C_M / GM_BM), blk, 0, stream>>>(
      AOb, Wob, nullptr, nullptr, out, nullptr, C_M, C_HIDDEN, C_HIDDEN);
}
